// ScrappedAttention_8263517077793
// MI455X (gfx1250) — matrix operand order repaired and checked by reading, not compiled, not run on hardware
//
#include <hip/hip_runtime.h>
#include <hip/hip_bf16.h>

// MI455X / gfx1250 attention layer:
//   QKV gemm (bf16 WMMA) -> flash attention (bf16 WMMA, fp32 softmax) -> out gemm (+bias)
// Matrix math: v_wmma_f32_16x16x32_bf16.  Data movement: global_load_async_to_lds_b128
// (CDNA5 async mem->LDS DMA, ASYNCcnt), double-buffered LDS, 2x-unrolled pipeline with
// loop-carried address registers (no per-iteration address rebuild).

typedef __bf16 bf16;
typedef __attribute__((ext_vector_type(16))) __bf16 v16bf;
typedef __attribute__((ext_vector_type(8)))  __bf16 v8bf;
typedef __attribute__((ext_vector_type(8)))  float  v8f;

union AFrag { v16bf v; v8bf h[2]; };   // A 16x32 bf16: two 16B chunks per lane

// softmax done in base-2 domain: score * F^-0.5 * log2(e), exp -> v_exp_f32 directly
static constexpr float SCL2 = 0.03125f * 1.4426950408889634f;

__device__ __forceinline__ v8f wmma_bf16(v16bf a, v16bf b, v8f c) {
  return __builtin_amdgcn_wmma_f32_16x16x32_bf16(false, a, false, b, (short)0, c,
                                                 false, false);
}

// CDNA5 async mem->LDS: 16 bytes per lane, no VGPR round trip, tracked by ASYNCcnt.
__device__ __forceinline__ void async_ld16(unsigned lds_off, const bf16* gsrc) {
  asm volatile("global_load_async_to_lds_b128 %0, %1, off"
               :: "v"(lds_off), "v"(gsrc)
               : "memory");
}
__device__ __forceinline__ void wait_async0() {
  asm volatile("s_wait_asynccnt 0" ::: "memory");
}
__device__ __forceinline__ unsigned lds_off(const bf16* p) {
  return (unsigned)(size_t)p;   // low 32 bits of generic LDS pointer = LDS byte offset
}

// ---------------------------------------------------------------------------
// fp32 -> bf16 conversion
// ---------------------------------------------------------------------------
__global__ void __launch_bounds__(256) cvt_f32_bf16(const float* __restrict__ s,
                                                    bf16* __restrict__ d, int n) {
  int i = blockIdx.x * 256 + threadIdx.x;
  if (i < n) d[i] = (bf16)s[i];
}

// ---------------------------------------------------------------------------
// Shared block-GEMM core: C(64x256) = A(64x1024) * B(Nx1024)^T, bf16 in, f32 acc.
// 8 waves as 2(M) x 4(N); each wave owns a 32x64 tile = 2x4 WMMA 16x16 tiles.
// Async double-buffered staging; pipeline unrolled by 2 so LDS offsets are constants.
// ---------------------------------------------------------------------------
__device__ __forceinline__ void gemm_tiles(const bf16* __restrict__ A,
                                           const bf16* __restrict__ Bm,
                                           int m0, int n0,
                                           bf16* lA2 /*2x64x32*/, bf16* lB2 /*2x256x32*/,
                                           v8f acc[2][4]) {
  constexpr int K = 1024;
  const int tid   = threadIdx.x;
  const int wave  = tid >> 5, lane = tid & 31;
  const int waveM = wave >> 2, waveN = wave & 3;
  const int half  = lane >> 4, l15 = lane & 15;

#pragma unroll
  for (int mi = 0; mi < 2; ++mi)
#pragma unroll
    for (int ni = 0; ni < 4; ++ni) acc[mi][ni] = (v8f){0.f,0.f,0.f,0.f,0.f,0.f,0.f,0.f};

  const int row = tid >> 2;          // 0..63
  const int col = (tid & 3) * 8;     // 0,8,16,24

  // Loop-carried global source pointers (point at most recently issued tile).
  const bf16* ap  = &A[(size_t)(m0 + row) * K + col];
  const bf16* bp0 = &Bm[(size_t)(n0 + row) * K + col];
  const bf16* bp1 = bp0 + (size_t)64 * K;
  const bf16* bp2 = bp0 + (size_t)128 * K;
  const bf16* bp3 = bp0 + (size_t)192 * K;
  // Per-thread LDS destination offsets (buffer 0); buffer strides are constants.
  const unsigned la0 = lds_off(&lA2[row * 32 + col]);
  const unsigned lb0 = lds_off(&lB2[row * 32 + col]);

  // Prologue: tile kk=0 -> buffer 0.
  async_ld16(la0, ap);
  async_ld16(lb0, bp0);
  async_ld16(lb0 + 4096, bp1);
  async_ld16(lb0 + 8192, bp2);
  async_ld16(lb0 + 12288, bp3);

  for (int kk = 0; kk < K; kk += 64) {
#pragma unroll
    for (int s = 0; s < 2; ++s) {          // s = current buffer (0,1)
      wait_async0();       // own async loads for buffer s have landed in LDS
      __syncthreads();     // everyone's loads visible; prev readers of buf s^1 done

      if (kk + 32 * s + 32 < K) {          // issue next tile -> buffer s^1
        const unsigned nb = (unsigned)(s ^ 1);
        ap += 32; bp0 += 32; bp1 += 32; bp2 += 32; bp3 += 32;
        async_ld16(la0 + nb * 4096, ap);
        async_ld16(lb0 + nb * 16384, bp0);
        async_ld16(lb0 + nb * 16384 + 4096, bp1);
        async_ld16(lb0 + nb * 16384 + 8192, bp2);
        async_ld16(lb0 + nb * 16384 + 12288, bp3);
      }

      const bf16* cA = lA2 + s * (64 * 32);
      const bf16* cB = lB2 + s * (256 * 32);

      AFrag a[2];
#pragma unroll
      for (int mi = 0; mi < 2; ++mi) {
        const bf16* base = &cA[(waveM * 32 + mi * 16 + l15) * 32];
        a[mi].h[0] = *(const v8bf*)&base[half * 16];
        a[mi].h[1] = *(const v8bf*)&base[half * 16 + 8];
      }
#pragma unroll
      for (int ni = 0; ni < 4; ++ni) {
        v16bf b = *(const v16bf*)&cB[(waveN * 64 + ni * 16 + l15) * 32 + half * 16];
#pragma unroll
        for (int mi = 0; mi < 2; ++mi)
          acc[mi][ni] = wmma_bf16(a[mi].v, b, acc[mi][ni]);
      }
    }
  }
}

// ---------------------------------------------------------------------------
// GEMM1: QKV = X * Wqkv^T, scattered to Q[B,H,T,DH], K[B,H,T,DH], Vt[B,H,DH,T]
// channel layout: o = d*48 + which*16 + h   (reshape (DH,3,H) in reference)
// ---------------------------------------------------------------------------
__global__ void __launch_bounds__(256) qkv_gemm(const bf16* __restrict__ xbf,
                                                const bf16* __restrict__ wq,
                                                bf16* __restrict__ qws,
                                                bf16* __restrict__ kws,
                                                bf16* __restrict__ vws) {
  __shared__ __align__(64) bf16 lA2[2 * 64 * 32];
  __shared__ __align__(64) bf16 lB2[2 * 256 * 32];
  const int m0 = blockIdx.x * 64;
  const int n0 = blockIdx.y * 256;
  v8f acc[2][4];
  gemm_tiles(xbf, wq, m0, n0, lA2, lB2, acc);

  const int tid  = threadIdx.x;
  const int wave = tid >> 5, lane = tid & 31;
  const int waveM = wave >> 2, waveN = wave & 3;
  const int half = lane >> 4, l15 = lane & 15;

#pragma unroll
  for (int mi = 0; mi < 2; ++mi) {
    const int mbase = m0 + waveM * 32 + mi * 16 + half * 8;
#pragma unroll
    for (int ni = 0; ni < 4; ++ni) {
      const int o     = n0 + waveN * 64 + ni * 16 + l15;
      const int d     = o / 48;
      const int rem   = o - d * 48;
      const int which = rem >> 4;   // uniform across the tile
      const int h     = rem & 15;
#pragma unroll
      for (int r = 0; r < 8; ++r) {
        const int m = mbase + r;
        const int b = m >> 11;         // / 2048
        const int t = m & 2047;
        const bf16 val = (bf16)acc[mi][ni][r];
        const size_t bh = (size_t)(b * 16 + h);
        if (which == 0)      qws[(bh * 2048 + t) * 64 + d] = val;
        else if (which == 1) kws[(bh * 2048 + t) * 64 + d] = val;
        else                 vws[(bh * 64 + d) * 2048 + t] = val;  // V transposed
      }
    }
  }
}

// ---------------------------------------------------------------------------
// Flash attention: block = 128 threads = 4 waves; each wave owns 16 query rows.
// K/V 32-key tiles double-buffered in LDS via async-to-LDS; base-2 online softmax.
// ---------------------------------------------------------------------------
__global__ void __launch_bounds__(128) attn_fwd(const bf16* __restrict__ q,
                                                const bf16* __restrict__ k,
                                                const bf16* __restrict__ vT,
                                                bf16* __restrict__ aout) {
  __shared__ __align__(64) bf16 lK2[2 * 32 * 64];   // 32 keys x 64 d, double buffered
  __shared__ __align__(64) bf16 lV2[2 * 64 * 32];   // 64 d x 32 keys (V^T)
  __shared__ __align__(64) bf16 lP[4][16 * 32];     // per-wave P staging

  const int bh  = blockIdx.x >> 5;   // (b*16 + h)
  const int rb  = blockIdx.x & 31;   // 64-row block within T
  const int tid = threadIdx.x;
  const int wave = tid >> 5, lane = tid & 31;
  const int half = lane >> 4, l15 = lane & 15;
  const int t0 = rb * 64 + wave * 16;

  const bf16* qb = q  + (size_t)bh * 2048 * 64;
  const bf16* kb = k  + (size_t)bh * 2048 * 64;
  const bf16* vb = vT + (size_t)bh * 64 * 2048;

  // Q A-fragments for the two K-steps of DH=64, kept in registers all loop.
  AFrag qf[2];
  {
    const bf16* qrow = qb + (size_t)(t0 + l15) * 64;
#pragma unroll
    for (int s = 0; s < 2; ++s) {
      qf[s].h[0] = *(const v8bf*)&qrow[s * 32 + half * 16];
      qf[s].h[1] = *(const v8bf*)&qrow[s * 32 + half * 16 + 8];
    }
  }

  v8f oacc[4];
#pragma unroll
  for (int n = 0; n < 4; ++n) oacc[n] = (v8f){0.f,0.f,0.f,0.f,0.f,0.f,0.f,0.f};
  float mrow[8], lrow[8];
#pragma unroll
  for (int r = 0; r < 8; ++r) { mrow[r] = -1e30f; lrow[r] = 0.f; }

  // Per-thread staging coordinates + loop-carried source pointers.
  const int i1 = tid + 128;
  const bf16* kp0 = &kb[(size_t)(tid >> 3) * 64 + (tid & 7) * 8];
  const bf16* kp1 = &kb[(size_t)(i1 >> 3) * 64 + (i1 & 7) * 8];
  const bf16* vp0 = &vb[(size_t)(tid >> 2) * 2048 + (tid & 3) * 8];
  const bf16* vp1 = &vb[(size_t)(i1 >> 2) * 2048 + (i1 & 3) * 8];
  const unsigned lk0 = lds_off(&lK2[(tid >> 3) * 64 + (tid & 7) * 8]);
  const unsigned lk1 = lds_off(&lK2[(i1 >> 3) * 64 + (i1 & 7) * 8]);
  const unsigned lv0 = lds_off(&lV2[(tid >> 2) * 32 + (tid & 3) * 8]);
  const unsigned lv1 = lds_off(&lV2[(i1 >> 2) * 32 + (i1 & 3) * 8]);

  // Prologue: key block jb=0 -> buffer 0.
  async_ld16(lk0, kp0);
  async_ld16(lk1, kp1);
  async_ld16(lv0, vp0);
  async_ld16(lv1, vp1);

  for (int jb = 0; jb < 2048; jb += 64) {
#pragma unroll
    for (int s = 0; s < 2; ++s) {          // s = current buffer (0,1)
      wait_async0();
      __syncthreads();

      if (jb + 32 * s + 32 < 2048) {       // issue next K/V tile -> buffer s^1
        const unsigned nb = (unsigned)(s ^ 1);
        kp0 += 32 * 64; kp1 += 32 * 64;    // advance 32 key rows
        vp0 += 32;      vp1 += 32;         // advance 32 key columns
        async_ld16(lk0 + nb * 4096, kp0);
        async_ld16(lk1 + nb * 4096, kp1);
        async_ld16(lv0 + nb * 4096, vp0);
        async_ld16(lv1 + nb * 4096, vp1);
      }

      const bf16* cK = lK2 + s * (32 * 64);
      const bf16* cV = lV2 + s * (64 * 32);

      // S(16x32) = Q(16x64) * K(32x64)^T  -> two 16x16 tiles, 2 K-steps each
      v8f s0 = (v8f){0.f,0.f,0.f,0.f,0.f,0.f,0.f,0.f};
      v8f s1 = (v8f){0.f,0.f,0.f,0.f,0.f,0.f,0.f,0.f};
#pragma unroll
      for (int st = 0; st < 2; ++st) {
        v16bf b0 = *(const v16bf*)&cK[(l15) * 64 + st * 32 + half * 16];
        v16bf b1 = *(const v16bf*)&cK[(16 + l15) * 64 + st * 32 + half * 16];
        s0 = wmma_bf16(qf[st].v, b0, s0);
        s1 = wmma_bf16(qf[st].v, b1, s1);
      }

      // Base-2 online softmax. Row = half*8 + r lives in 16 lanes of one half.
      float alpha[8];
      bf16* pl = lP[wave];
#pragma unroll
      for (int r = 0; r < 8; ++r) {
        const float a0 = s0[r] * SCL2, a1 = s1[r] * SCL2;
        float mx = fmaxf(a0, a1);
#pragma unroll
        for (int off = 1; off < 16; off <<= 1) mx = fmaxf(mx, __shfl_xor(mx, off, 16));
        const float mnew = fmaxf(mrow[r], mx);
        const float al = exp2f(mrow[r] - mnew);
        const float e0 = exp2f(a0 - mnew);
        const float e1 = exp2f(a1 - mnew);
        float sum = e0 + e1;
#pragma unroll
        for (int off = 1; off < 16; off <<= 1) sum += __shfl_xor(sum, off, 16);
        lrow[r] = lrow[r] * al + sum;
        mrow[r] = mnew;
        alpha[r] = al;
        // P (C-layout) -> per-wave LDS, row-major 16x32 bf16
        pl[(half * 8 + r) * 32 + l15]      = (bf16)e0;
        pl[(half * 8 + r) * 32 + 16 + l15] = (bf16)e1;
      }
#pragma unroll
      for (int n = 0; n < 4; ++n)
#pragma unroll
        for (int r = 0; r < 8; ++r) oacc[n][r] *= alpha[r];

      // CDNA5 split-counter wait: P stores must land before A-layout reload.
      asm volatile("s_wait_dscnt 0" ::: "memory");

      AFrag pf;  // P as A-fragment (16x32)
      const bf16* prow = &pl[l15 * 32];
      pf.h[0] = *(const v8bf*)&prow[half * 16];
      pf.h[1] = *(const v8bf*)&prow[half * 16 + 8];

      // O(16x64) += P(16x32) * V(32x64)
#pragma unroll
      for (int n = 0; n < 4; ++n) {
        v16bf bv = *(const v16bf*)&cV[(n * 16 + l15) * 32 + half * 16];
        oacc[n] = wmma_bf16(pf.v, bv, oacc[n]);
      }
    }
  }

  float inv[8];
#pragma unroll
  for (int r = 0; r < 8; ++r) inv[r] = 1.0f / lrow[r];

  const int b = bh >> 4, h = bh & 15;
#pragma unroll
  for (int n = 0; n < 4; ++n) {
    const int d = n * 16 + l15;
#pragma unroll
    for (int r = 0; r < 8; ++r) {
      const int t = t0 + half * 8 + r;
      aout[((size_t)(b * 2048 + t)) * 1024 + h * 64 + d] = (bf16)(oacc[n][r] * inv[r]);
    }
  }
}

// ---------------------------------------------------------------------------
// GEMM2: out = Attn * Wout^T + bias  (fp32 output)
// ---------------------------------------------------------------------------
__global__ void __launch_bounds__(256) out_gemm(const bf16* __restrict__ attn,
                                                const bf16* __restrict__ wo,
                                                const float* __restrict__ bias,
                                                float* __restrict__ out) {
  __shared__ __align__(64) bf16 lA2[2 * 64 * 32];
  __shared__ __align__(64) bf16 lB2[2 * 256 * 32];
  const int m0 = blockIdx.x * 64;
  const int n0 = blockIdx.y * 256;
  v8f acc[2][4];
  gemm_tiles(attn, wo, m0, n0, lA2, lB2, acc);

  const int tid  = threadIdx.x;
  const int wave = tid >> 5, lane = tid & 31;
  const int waveM = wave >> 2, waveN = wave & 3;
  const int half = lane >> 4, l15 = lane & 15;

#pragma unroll
  for (int ni = 0; ni < 4; ++ni) {
    const int o = n0 + waveN * 64 + ni * 16 + l15;
    const float bo = bias[o];
#pragma unroll
    for (int mi = 0; mi < 2; ++mi) {
      const int mbase = m0 + waveM * 32 + mi * 16 + half * 8;
#pragma unroll
      for (int r = 0; r < 8; ++r)
        out[(size_t)(mbase + r) * 1024 + o] = acc[mi][ni][r] + bo;
    }
  }
}

// ---------------------------------------------------------------------------
extern "C" void kernel_launch(void* const* d_in, const int* in_sizes, int n_in,
                              void* d_out, int out_size, void* d_ws, size_t ws_size,
                              hipStream_t stream) {
  (void)in_sizes; (void)n_in; (void)out_size; (void)ws_size;
  const float* x     = (const float*)d_in[0];   // (4,2048,1024)
  const float* w_qkv = (const float*)d_in[1];   // (3072,1024)
  const float* w_out = (const float*)d_in[2];   // (1024,1024)
  const float* b_out = (const float*)d_in[3];   // (1024,)
  float* out = (float*)d_out;                   // (4,2048,1024)

  char* ws = (char*)d_ws;
  size_t off = 0;
  auto salloc = [&](size_t bytes) -> void* {
    void* p = ws + off;
    off += (bytes + 255) / 256 * 256;
    return p;
  };
  bf16* xbf  = (bf16*)salloc((size_t)8192 * 1024 * 2);      // 16 MB
  bf16* wqbf = (bf16*)salloc((size_t)3072 * 1024 * 2);      //  6 MB
  bf16* wobf = (bf16*)salloc((size_t)1024 * 1024 * 2);      //  2 MB
  bf16* qws  = (bf16*)salloc((size_t)64 * 2048 * 64 * 2);   // 16 MB  [B*H,T,DH]
  bf16* kws  = (bf16*)salloc((size_t)64 * 2048 * 64 * 2);   // 16 MB  [B*H,T,DH]
  bf16* vws  = (bf16*)salloc((size_t)64 * 64 * 2048 * 2);   // 16 MB  [B*H,DH,T]
  bf16* aws  = (bf16*)salloc((size_t)8192 * 1024 * 2);      // 16 MB  (B*T, H*DH)

  cvt_f32_bf16<<<(8388608 + 255) / 256, 256, 0, stream>>>(x, xbf, 8388608);
  cvt_f32_bf16<<<(3145728 + 255) / 256, 256, 0, stream>>>(w_qkv, wqbf, 3145728);
  cvt_f32_bf16<<<(1048576 + 255) / 256, 256, 0, stream>>>(w_out, wobf, 1048576);

  qkv_gemm<<<dim3(128, 12), 256, 0, stream>>>(xbf, wqbf, qws, kws, vws);
  attn_fwd<<<2048, 128, 0, stream>>>(qws, kws, vws, aws);
  out_gemm<<<dim3(128, 4), 256, 0, stream>>>(aws, wobf, b_out, out);
}
